// CrystalGCNEncoder_16260746183170
// MI455X (gfx1250) — compile-verified, matrix operand order checked by reading, not run on hardware
//
#include <hip/hip_runtime.h>

// ---------------------------------------------------------------------------
// CrystalGCN encoder for MI455X (gfx1250): all dense GEMMs through
// v_wmma_f32_16x16x32_bf16 (bf16 inputs, f32 accumulate) with LDS-staged,
// double-buffered B tiles (true ds_load/ds_store path: buffer selection via
// integer offsets, never via generic pointers) and register-prefetched A.
// Graph-irregular parts (segment softmax over edges, per-graph attention
// pooling) via float atomics. wave32 throughout.
// ---------------------------------------------------------------------------

typedef unsigned short u16;
typedef __attribute__((ext_vector_type(16))) __bf16 bf16x16;
typedef __attribute__((ext_vector_type(8)))  float  floatx8;

#define TB 256
#define NHEAD 4
#define BSTRIDE 40               // u16 per staged B row (80 B: conflict-free)
#define BUFSZ (64 * BSTRIDE)     // u16 per LDS buffer (5120 B)

static __device__ __forceinline__ u16 f2bf(float f) {
  unsigned int u = __float_as_uint(f);
  unsigned int r = 0x7FFFu + ((u >> 16) & 1u);   // round-to-nearest-even
  return (u16)((u + r) >> 16);
}

static __device__ __forceinline__ void atomicMaxF(float* addr, float v) {
  if (v >= 0.0f) atomicMax((int*)addr, __float_as_int(v));
  else           atomicMin((unsigned int*)addr, __float_as_uint(v));
}

// --------------------------- elementwise utils -----------------------------

__global__ void k_fill(float* __restrict__ p, float v, long long n) {
  long long i = (long long)blockIdx.x * blockDim.x + threadIdx.x;
  if (i < n) p[i] = v;
}

__global__ void k_cvt_bf16(const float* __restrict__ s, u16* __restrict__ d,
                           long long n) {
  long long i = (long long)blockIdx.x * blockDim.x + threadIdx.x;
  if (i < n) d[i] = f2bf(s[i]);
}

// W [K, Nout] f32 row-major  ->  Wt [Nout, K] bf16 row-major (col-major W)
__global__ void k_transpose_bf16(const float* __restrict__ W,
                                 u16* __restrict__ Wt, int K, int Nout) {
  long long i = (long long)blockIdx.x * blockDim.x + threadIdx.x;
  if (i >= (long long)K * Nout) return;
  int n = (int)(i % Nout);
  int k = (int)(i / Nout);
  Wt[(size_t)n * K + k] = f2bf(W[i]);
}

// ------------------------------- WMMA GEMM ---------------------------------
// C[M, Nout] = A_bf16[M, K] * Bt_bf16[Nout, K]^T (+ bias).
// Block = 8 waves = 128x64 C tile. The 64x32 B chunk for each K-step is
// loaded once per block, double-buffered in LDS (padded stride, conflict
// free), and shared by all 8 waves; A fragments are register double-buffered
// per wave. Chunk i+2 is in flight from global while chunk i is consumed.
// LDS buffer selection is an integer element offset so every access stays a
// GEP on the __shared__ array (addrspace(3) -> real ds_load/ds_store, not
// flat). M multiple of 16, K multiple of 32, Nout multiple of 64. Waves whose
// row tile exceeds M are clamped (redundant compute) so barriers stay
// uniform; rows >= Mstore are never stored.

__global__ __launch_bounds__(TB)
void k_gemm_wmma(const u16* __restrict__ A, const u16* __restrict__ Bt,
                 const float* __restrict__ bias, float* __restrict__ C,
                 int M, int Mstore, int K, int Nout) {
  __shared__ u16 smem[2 * BUFSZ];                  // 10240 B

  const int lane   = threadIdx.x & 31;
  const int wave   = threadIdx.x >> 5;
  const int tilesN = Nout >> 6;
  const int tilesM = M >> 4;
  const int tmb    = blockIdx.x / tilesN;
  const int tn     = blockIdx.x % tilesN;
  int tm = tmb * 8 + wave;
  if (tm >= tilesM) tm = tilesM - 1;               // clamp, store-guarded

  const int laneLo = lane & 15;
  const int kHalf  = (lane >> 4) * 16;             // 0 or 16 (k elements)

  const u16* aptr = A + (size_t)(tm * 16 + laneLo) * K + kHalf;

  // cooperative B staging: thread t moves 16 B of Bt row (tn*64 + c)
  const int  c   = threadIdx.x >> 2;               // 0..63 (output column)
  const int  s   = threadIdx.x & 3;                // 16B segment in 64B row
  const u16* gB  = Bt + (size_t)(tn * 64 + c) * K + s * 8;
  const int sOff = c * BSTRIDE + s * 8;            // element offset in buffer

  const int nk = K >> 5;

  // ---- prologue: chunk0 -> LDS buf0, chunk1 -> regs, A chunk0 -> regs ----
  uint4 breg = *reinterpret_cast<const uint4*>(gB);
  bf16x16 a0;
  __builtin_memcpy(&a0, aptr, 32);
  __builtin_memcpy(&smem[sOff], &breg, 16);
  if (nk > 1) breg = *reinterpret_cast<const uint4*>(gB + 32);
  __syncthreads();

  floatx8 acc0 = {}, acc1 = {}, acc2 = {}, acc3 = {};
  int base = 0;                                    // 0 <-> BUFSZ toggle

  for (int i = 0; i < nk; ++i) {
    bf16x16 a1;
    if (i + 1 < nk) __builtin_memcpy(&a1, aptr + (size_t)(i + 1) * 32, 32);

    const int bb = base + kHalf;
    bf16x16 b0, b1, b2, b3;
    __builtin_memcpy(&b0, &smem[bb + (laneLo +  0) * BSTRIDE], 32);
    __builtin_memcpy(&b1, &smem[bb + (laneLo + 16) * BSTRIDE], 32);
    __builtin_memcpy(&b2, &smem[bb + (laneLo + 32) * BSTRIDE], 32);
    __builtin_memcpy(&b3, &smem[bb + (laneLo + 48) * BSTRIDE], 32);

    acc0 = __builtin_amdgcn_wmma_f32_16x16x32_bf16(false, a0, false, b0,
                                                   (short)0, acc0, false, false);
    acc1 = __builtin_amdgcn_wmma_f32_16x16x32_bf16(false, a0, false, b1,
                                                   (short)0, acc1, false, false);
    acc2 = __builtin_amdgcn_wmma_f32_16x16x32_bf16(false, a0, false, b2,
                                                   (short)0, acc2, false, false);
    acc3 = __builtin_amdgcn_wmma_f32_16x16x32_bf16(false, a0, false, b3,
                                                   (short)0, acc3, false, false);

    if (i + 1 < nk) {
      uint4 w = breg;                                        // chunk i+1
      if (i + 2 < nk)                                        // chunk i+2 ->
        breg = *reinterpret_cast<const uint4*>(gB + (size_t)(i + 2) * 32);
      __syncthreads();               // readers of other buffer (iter i-1) done
      __builtin_memcpy(&smem[(BUFSZ - base) + sOff], &w, 16);
      __syncthreads();               // chunk i+1 visible to all waves
      base = BUFSZ - base;
      a0 = a1;
    }
  }

  // C/D layout: VGPR r -> row r (lanes 0-15) / row r+8 (lanes 16-31)
  const int rbase = tm * 16 + ((lane >> 4) ? 8 : 0);
  floatx8 accs[4] = {acc0, acc1, acc2, acc3};
#pragma unroll
  for (int j = 0; j < 4; ++j) {
    const int col = tn * 64 + j * 16 + laneLo;
    const float bv = bias ? bias[col] : 0.0f;
#pragma unroll
    for (int r = 0; r < 8; ++r) {
      const int row = rbase + r;
      if (row < Mstore) C[(size_t)row * Nout + col] = accs[j][r] + bv;
    }
  }
}

// ----------------------------- GAT attention -------------------------------

__global__ void k_el_er(const float* __restrict__ feat,
                        const float* __restrict__ al,
                        const float* __restrict__ ar,
                        float* __restrict__ el, float* __restrict__ er,
                        int N, int O) {
  int i = blockIdx.x * blockDim.x + threadIdx.x;          // n*NHEAD + h
  if (i >= N * NHEAD) return;
  int h = i & (NHEAD - 1);
  int n = i >> 2;
  const float* f = feat + (size_t)n * NHEAD * O + (size_t)h * O;
  const float* a = al + (size_t)h * O;
  const float* b = ar + (size_t)h * O;
  float sl = 0.0f, sr = 0.0f;
  for (int o = 0; o < O; ++o) { float v = f[o]; sl += v * a[o]; sr += v * b[o]; }
  el[i] = sl; er[i] = sr;
}

__global__ void k_edge_max(const int* __restrict__ src, const int* __restrict__ dst,
                           const float* __restrict__ el, const float* __restrict__ er,
                           float* __restrict__ ebuf, float* __restrict__ emax, int E) {
  int i = blockIdx.x * blockDim.x + threadIdx.x;          // e*NHEAD + h
  if (i >= E * NHEAD) return;
  int h = i & (NHEAD - 1);
  int e = i >> 2;
  float x = el[src[e] * NHEAD + h] + er[dst[e] * NHEAD + h];
  x = (x > 0.0f) ? x : 0.2f * x;
  ebuf[i] = x;
  atomicMaxF(&emax[dst[e] * NHEAD + h], x);
}

__global__ void k_edge_exp(const int* __restrict__ dst, float* __restrict__ ebuf,
                           const float* __restrict__ emax, float* __restrict__ denom,
                           int E) {
  int i = blockIdx.x * blockDim.x + threadIdx.x;
  if (i >= E * NHEAD) return;
  int h = i & (NHEAD - 1);
  int e = i >> 2;
  float ex = __expf(ebuf[i] - emax[dst[e] * NHEAD + h]);
  ebuf[i] = ex;
  atomicAdd(&denom[dst[e] * NHEAD + h], ex);
}

__global__ void k_edge_scatter(const int* __restrict__ src, const int* __restrict__ dst,
                               const float* __restrict__ feat,
                               const float* __restrict__ ebuf,
                               const float* __restrict__ denom,
                               float* __restrict__ rst, int E, int O) {
  long long i = (long long)blockIdx.x * blockDim.x + threadIdx.x;
  const int chunks = O >> 2;
  if (i >= (long long)E * NHEAD * chunks) return;
  int c = (int)(i % chunks);
  int t = (int)(i / chunks);
  int h = t & (NHEAD - 1);
  int e = t >> 2;
  const int D = NHEAD * O;
  float alpha = ebuf[t] / fmaxf(denom[dst[e] * NHEAD + h], 1e-9f);
  const float4 f = *reinterpret_cast<const float4*>(
      feat + (size_t)src[e] * D + (size_t)h * O + (size_t)c * 4);
  float* r = rst + (size_t)dst[e] * D + (size_t)h * O + (size_t)c * 4;
  atomicAdd(r + 0, f.x * alpha);
  atomicAdd(r + 1, f.y * alpha);
  atomicAdd(r + 2, f.z * alpha);
  atomicAdd(r + 3, f.w * alpha);
}

__global__ void k_bias_elu(const float* __restrict__ rst, const float* __restrict__ b,
                           float* __restrict__ hout, u16* __restrict__ hbf,
                           int N, int D) {
  long long i = (long long)blockIdx.x * blockDim.x + threadIdx.x;
  if (i >= (long long)N * D) return;
  int d = (int)(i % D);
  float v = rst[i] + b[d];
  v = (v > 0.0f) ? v : expm1f(v);
  hout[i] = v;
  hbf[i] = f2bf(v);
}

// ------------------------- global attention pooling ------------------------

__global__ void k_gate(const float* __restrict__ g1, const float* __restrict__ Wg2,
                       const float* __restrict__ bg2, float* __restrict__ gate, int N) {
  int n = blockIdx.x * blockDim.x + threadIdx.x;
  if (n >= N) return;
  const float* g = g1 + (size_t)n * 128;
  float s = bg2[0];
  for (int j = 0; j < 128; ++j) { float v = g[j]; v = (v > 0.0f) ? v : 0.0f; s += v * Wg2[j]; }
  gate[n] = s;
}

__global__ void k_graph_max(const int* __restrict__ gid, const float* __restrict__ gate,
                            float* __restrict__ gmax, int N) {
  int n = blockIdx.x * blockDim.x + threadIdx.x;
  if (n >= N) return;
  atomicMaxF(&gmax[gid[n]], gate[n]);
}

__global__ void k_graph_expsum(const int* __restrict__ gid, float* __restrict__ gate,
                               const float* __restrict__ gmax, float* __restrict__ gsum,
                               int N) {
  int n = blockIdx.x * blockDim.x + threadIdx.x;
  if (n >= N) return;
  float e = __expf(gate[n] - gmax[gid[n]]);
  gate[n] = e;
  atomicAdd(&gsum[gid[n]], e);
}

__global__ void k_pool(const int* __restrict__ gid, const float* __restrict__ gate,
                       const float* __restrict__ gsum, const float* __restrict__ h3,
                       float* __restrict__ gemb, int N) {
  const int D = 1024, chunks = D / 4;
  long long i = (long long)blockIdx.x * blockDim.x + threadIdx.x;
  if (i >= (long long)N * chunks) return;
  int c = (int)(i % chunks);
  int n = (int)(i / chunks);
  int g = gid[n];
  float a = gate[n] / fmaxf(gsum[g], 1e-9f);
  const float4 hv = *reinterpret_cast<const float4*>(h3 + (size_t)n * D + (size_t)c * 4);
  float* p = gemb + (size_t)g * D + (size_t)c * 4;
  atomicAdd(p + 0, hv.x * a);
  atomicAdd(p + 1, hv.y * a);
  atomicAdd(p + 2, hv.z * a);
  atomicAdd(p + 3, hv.w * a);
}

// ------------------------------- launcher ----------------------------------

extern "C" void kernel_launch(void* const* d_in, const int* in_sizes, int n_in,
                              void* d_out, int out_size, void* d_ws, size_t ws_size,
                              hipStream_t stream) {
  const float* node_feat = (const float*)d_in[0];
  const int*   src  = (const int*)d_in[1];
  const int*   dst  = (const int*)d_in[2];
  const int*   gid  = (const int*)d_in[3];
  const float* W[3]  = {(const float*)d_in[4],  (const float*)d_in[8],  (const float*)d_in[12]};
  const float* al[3] = {(const float*)d_in[5],  (const float*)d_in[9],  (const float*)d_in[13]};
  const float* ar[3] = {(const float*)d_in[6],  (const float*)d_in[10], (const float*)d_in[14]};
  const float* bL[3] = {(const float*)d_in[7],  (const float*)d_in[11], (const float*)d_in[15]};
  const float* Wg1 = (const float*)d_in[16];
  const float* bg1 = (const float*)d_in[17];
  const float* Wg2 = (const float*)d_in[18];
  const float* bg2 = (const float*)d_in[19];
  const float* Wmu = (const float*)d_in[20];
  const float* bmu = (const float*)d_in[21];
  const float* Wlv = (const float*)d_in[22];
  const float* blv = (const float*)d_in[23];

  const int N = in_sizes[0] / 128;           // 20000
  const int E = in_sizes[1];                 // 320000
  const int G = out_size / (2 * 128);        // 200
  const int Gp = ((G + 15) / 16) * 16;       // 208 (padded row count)
  const int Kin[3]  = {128, 256, 512};
  const int Dn[3]   = {256, 512, 1024};
  const int Ov[3]   = {64, 128, 256};
  const float NEG_INF = -__builtin_huge_valf();

  // ------- workspace carve-up (256B aligned) -------
  char* ws = (char*)d_ws;
  size_t off = 0;
  auto take = [&](size_t bytes) -> char* {
    char* p = ws + off;
    off += (bytes + 255) & ~(size_t)255;
    return p;
  };
  u16*   hbf    = (u16*)  take((size_t)N * 1024 * sizeof(u16));   // activation bf16
  float* featb  = (float*)take((size_t)N * 1024 * sizeof(float)); // feat, then h
  float* rstb   = (float*)take((size_t)N * 1024 * sizeof(float));
  u16*   W1t    = (u16*)  take((size_t)256  * 128  * sizeof(u16));
  u16*   W2t    = (u16*)  take((size_t)512  * 256  * sizeof(u16));
  u16*   W3t    = (u16*)  take((size_t)1024 * 512  * sizeof(u16));
  u16*   Wg1t   = (u16*)  take((size_t)128  * 1024 * sizeof(u16));
  u16*   Wmut   = (u16*)  take((size_t)128  * 1024 * sizeof(u16));
  u16*   Wlvt   = (u16*)  take((size_t)128  * 1024 * sizeof(u16));
  u16*   Wt[3]  = {W1t, W2t, W3t};
  float* el     = (float*)take((size_t)N * NHEAD * sizeof(float));
  float* er     = (float*)take((size_t)N * NHEAD * sizeof(float));
  float* emax   = (float*)take((size_t)N * NHEAD * sizeof(float));
  float* denom  = (float*)take((size_t)N * NHEAD * sizeof(float));
  float* ebuf   = (float*)take((size_t)E * NHEAD * sizeof(float));
  float* g1     = (float*)take((size_t)N * 128 * sizeof(float));
  float* gate   = (float*)take((size_t)N * sizeof(float));
  float* gmax   = (float*)take((size_t)G * sizeof(float));
  float* gsum   = (float*)take((size_t)G * sizeof(float));
  float* gemb   = (float*)take((size_t)Gp * 1024 * sizeof(float));
  u16*   gembbf = (u16*)  take((size_t)Gp * 1024 * sizeof(u16));
  (void)ws_size; (void)n_in;

  auto nb = [](long long n) { return (int)((n + TB - 1) / TB); };
  auto gemmBlocks = [](int M, int Nout) {
    int tm = M / 16;
    int bm = (tm + 7) / 8;                   // 8 row-tiles per block
    return bm * (Nout / 64);
  };

  // ------- weight prep (bf16, transposed) + input convert -------
  k_transpose_bf16<<<nb(128 * 256),   TB, 0, stream>>>(W[0], W1t, 128, 256);
  k_transpose_bf16<<<nb(256 * 512),   TB, 0, stream>>>(W[1], W2t, 256, 512);
  k_transpose_bf16<<<nb(512 * 1024),  TB, 0, stream>>>(W[2], W3t, 512, 1024);
  k_transpose_bf16<<<nb(1024 * 128),  TB, 0, stream>>>(Wg1, Wg1t, 1024, 128);
  k_transpose_bf16<<<nb(1024 * 128),  TB, 0, stream>>>(Wmu, Wmut, 1024, 128);
  k_transpose_bf16<<<nb(1024 * 128),  TB, 0, stream>>>(Wlv, Wlvt, 1024, 128);
  k_cvt_bf16<<<nb((long long)N * 128), TB, 0, stream>>>(node_feat, hbf, (long long)N * 128);

  // ------- 3 GAT layers -------
  for (int L = 0; L < 3; ++L) {
    const int K = Kin[L], D = Dn[L], O = Ov[L];
    k_gemm_wmma<<<gemmBlocks(N, D), TB, 0, stream>>>(hbf, Wt[L], nullptr, featb, N, N, K, D);
    k_el_er<<<nb((long long)N * NHEAD), TB, 0, stream>>>(featb, al[L], ar[L], el, er, N, O);
    k_fill<<<nb((long long)N * NHEAD), TB, 0, stream>>>(emax, NEG_INF, (long long)N * NHEAD);
    k_fill<<<nb((long long)N * NHEAD), TB, 0, stream>>>(denom, 0.0f, (long long)N * NHEAD);
    k_fill<<<nb((long long)N * D), TB, 0, stream>>>(rstb, 0.0f, (long long)N * D);
    k_edge_max<<<nb((long long)E * NHEAD), TB, 0, stream>>>(src, dst, el, er, ebuf, emax, E);
    k_edge_exp<<<nb((long long)E * NHEAD), TB, 0, stream>>>(dst, ebuf, emax, denom, E);
    k_edge_scatter<<<nb((long long)E * NHEAD * (O / 4)), TB, 0, stream>>>(
        src, dst, featb, ebuf, denom, rstb, E, O);
    k_bias_elu<<<nb((long long)N * D), TB, 0, stream>>>(rstb, bL[L], featb, hbf, N, D);
  }
  // featb now holds h3 (f32), hbf holds h3 (bf16)

  // ------- global attention pooling -------
  k_gemm_wmma<<<gemmBlocks(N, 128), TB, 0, stream>>>(hbf, Wg1t, bg1, g1, N, N, 1024, 128);
  k_gate<<<nb(N), TB, 0, stream>>>(g1, Wg2, bg2, gate, N);
  k_fill<<<nb(G), TB, 0, stream>>>(gmax, NEG_INF, G);
  k_fill<<<nb(G), TB, 0, stream>>>(gsum, 0.0f, G);
  k_graph_max<<<nb(N), TB, 0, stream>>>(gid, gate, gmax, N);
  k_graph_expsum<<<nb(N), TB, 0, stream>>>(gid, gate, gmax, gsum, N);
  k_fill<<<nb((long long)Gp * 1024), TB, 0, stream>>>(gemb, 0.0f, (long long)Gp * 1024);
  k_pool<<<nb((long long)N * 256), TB, 0, stream>>>(gid, gate, gsum, featb, gemb, N);
  k_cvt_bf16<<<nb((long long)G * 1024), TB, 0, stream>>>(gemb, gembbf, (long long)G * 1024);

  // ------- latent heads: mu / logvar -------
  float* out = (float*)d_out;
  k_gemm_wmma<<<gemmBlocks(Gp, 128), TB, 0, stream>>>(gembbf, Wmut, bmu, out, Gp, G, 1024, 128);
  k_gemm_wmma<<<gemmBlocks(Gp, 128), TB, 0, stream>>>(gembbf, Wlvt, blv,
                                                      out + (size_t)G * 128, Gp, G, 1024, 128);
}
